// VSSBlock_88510686036383
// MI455X (gfx1250) — compile-verified, matrix operand order checked by reading, not run on hardware
//
#include <hip/hip_runtime.h>
#include <hip/hip_bf16.h>
#include <math.h>

typedef __attribute__((ext_vector_type(16))) _Float16 v16h;
typedef __attribute__((ext_vector_type(8)))  _Float16 v8h;
typedef __attribute__((ext_vector_type(4)))  _Float16 v4h;
typedef __attribute__((ext_vector_type(8)))  float    v8f;
typedef __attribute__((ext_vector_type(4)))  int      v4i;

#define BATCH   8
#define CCH     384
#define HW      1024          // 32*32
#define MROWS   8192          // B * L
#define DINNER  768
#define DSTATE  16
#define DTRANK  24
#define LDSSTR  48            // LDS row stride in halves (96B, keeps b128 16B-aligned)

// Async global->LDS DMA path (gfx1250): guarded so the file still compiles if
// this toolchain does not declare the builtins; fallback = reg-staged copy.
#if defined(__AMDGCN__) &&                                                  \
    __has_builtin(__builtin_amdgcn_global_load_async_to_lds_b128) &&        \
    __has_builtin(__builtin_amdgcn_s_wait_asynccnt)
#define ASYNC_LDS 1
#define GLOBAL_AS __attribute__((address_space(1)))
#define LDS_AS    __attribute__((address_space(3)))
#else
#define ASYNC_LDS 0
#endif

// ---------------------------------------------------------------------------
// WMMA fragment load from LDS (16-bit A/B layout, cdna5 CDNA5 ISA 7.12.2):
// lane l -> row l&15;  halves j<8: K = 8*(l>>4)+j ; j>=8: K = 16+8*(l>>4)+(j-8)
// ---------------------------------------------------------------------------
__device__ inline v16h load_frag16(const _Float16* base, int lane) {
    const _Float16* p = base + (lane & 15) * LDSSTR + (lane >> 4) * 8;
    v8h lo = *(const v8h*)p;          // ds_load_b128
    v8h hi = *(const v8h*)(p + 16);   // ds_load_b128
    v16h r;
#pragma unroll
    for (int i = 0; i < 8; ++i) { r[i] = lo[i]; r[i + 8] = hi[i]; }
    return r;
}

__device__ inline v8f wmma32(v16h a, v16h b, v8f c) {
    return __builtin_amdgcn_wmma_f32_16x16x32_f16(false, a, false, b,
                                                  (short)0, c, false, false);
}

// ---------------------------------------------------------------------------
// Shared WMMA GEMM: out(M,N) = A(M,K) * W(N,K)^T ; A,W are f16 row-major.
// Block tile 64(M) x 64(N); 128 threads = 4 waves; each wave owns a 32x32
// sub-tile = 2x2 WMMA frags (4 v_wmma per K-step, fragment reuse).
// Double-buffered LDS; next tile staged via async global->LDS DMA when
// available (ASYNCcnt), else via registers. 1 barrier per K-step.
// ---------------------------------------------------------------------------
enum { EPI_F16 = 0, EPI_XPROJ = 1, EPI_SOFTPLUS = 2, EPI_GELU_F16 = 3, EPI_TRANS = 4 };

template <int EPI>
__global__ __launch_bounds__(128) void wmma_gemm(
    const _Float16* __restrict__ A, int lda,
    const _Float16* __restrict__ W, int ldw,
    int K,
    float* __restrict__ outF, _Float16* __restrict__ outH, int ldo,
    const float* __restrict__ bias,
    _Float16* __restrict__ dtA, float* __restrict__ BmO, float* __restrict__ CmO)
{
    const int lane = threadIdx.x & 31;
    const int wv   = threadIdx.x >> 5;        // 0..3
    const int m0   = blockIdx.y * 64;
    const int n0   = blockIdx.x * 64;
    const int wmi  = (wv & 1) * 32;           // wave's M sub-tile (32 rows)
    const int wni  = (wv >> 1) * 32;          // wave's N sub-tile (32 cols)

    __shared__ __align__(16) _Float16 sA[2][64][LDSSTR];
    __shared__ __align__(16) _Float16 sW[2][64][LDSSTR];

    v8f c00 = {}, c01 = {}, c10 = {}, c11 = {};

    // cooperative staging: 128 threads, 64 rows x 32 halves per tile
    // thread t: row = t>>1, col0 = (t&1)*16 ; two b128 (16B) chunks per tile
    const int grow = threadIdx.x >> 1;
    const int gcol = (threadIdx.x & 1) * 16;
    const _Float16* gA = A + (size_t)(m0 + grow) * lda + gcol;
    const _Float16* gW = W + (size_t)(n0 + grow) * ldw + gcol;

#if ASYNC_LDS
    // issue one K-step's 4 async b128 DMAs (this thread's share of both tiles)
    auto stage_async = [&](int kk, int nb) {
        __builtin_amdgcn_global_load_async_to_lds_b128(
            (GLOBAL_AS v4i*)(gA + kk),
            (LDS_AS v4i*)&sA[nb][grow][gcol], 0, 0);
        __builtin_amdgcn_global_load_async_to_lds_b128(
            (GLOBAL_AS v4i*)(gA + kk + 8),
            (LDS_AS v4i*)&sA[nb][grow][gcol + 8], 0, 0);
        __builtin_amdgcn_global_load_async_to_lds_b128(
            (GLOBAL_AS v4i*)(gW + kk),
            (LDS_AS v4i*)&sW[nb][grow][gcol], 0, 0);
        __builtin_amdgcn_global_load_async_to_lds_b128(
            (GLOBAL_AS v4i*)(gW + kk + 8),
            (LDS_AS v4i*)&sW[nb][grow][gcol + 8], 0, 0);
    };
    // ---- prologue: DMA k0 = 0 into buffer 0
    stage_async(0, 0);
    __builtin_amdgcn_s_wait_asynccnt(0);
#else
    {   // ---- prologue: stage k0 = 0 into LDS buffer 0 via registers
        v8h a0 = *(const v8h*)(gA);
        v8h a1 = *(const v8h*)(gA + 8);
        v8h w0 = *(const v8h*)(gW);
        v8h w1 = *(const v8h*)(gW + 8);
        *(v8h*)&sA[0][grow][gcol]     = a0;
        *(v8h*)&sA[0][grow][gcol + 8] = a1;
        *(v8h*)&sW[0][grow][gcol]     = w0;
        *(v8h*)&sW[0][grow][gcol + 8] = w1;
    }
#endif
    __syncthreads();

    int buf = 0;
    for (int k0 = 0; k0 < K; k0 += 32) {
        const bool has_next = (k0 + 32) < K;
#if ASYNC_LDS
        if (has_next) {
            stage_async(k0 + 32, buf ^ 1);       // DMA overlaps the WMMAs below
            __builtin_prefetch(gA + k0 + 64, 0, 1);
            __builtin_prefetch(gW + k0 + 64, 0, 1);
        }
#else
        v8h a0, a1, w0, w1;
        if (has_next) {
            a0 = *(const v8h*)(gA + k0 + 32);
            a1 = *(const v8h*)(gA + k0 + 40);
            w0 = *(const v8h*)(gW + k0 + 32);
            w1 = *(const v8h*)(gW + k0 + 40);
            __builtin_prefetch(gA + k0 + 64, 0, 1);
            __builtin_prefetch(gW + k0 + 64, 0, 1);
        }
#endif
        // compute current K-step from LDS[buf]
        v16h af0 = load_frag16(&sA[buf][wmi][0],      lane);
        v16h af1 = load_frag16(&sA[buf][wmi + 16][0], lane);
        v16h wf0 = load_frag16(&sW[buf][wni][0],      lane);
        v16h wf1 = load_frag16(&sW[buf][wni + 16][0], lane);
        c00 = wmma32(af0, wf0, c00);
        c01 = wmma32(af0, wf1, c01);
        c10 = wmma32(af1, wf0, c10);
        c11 = wmma32(af1, wf1, c11);

        if (has_next) {
#if ASYNC_LDS
            __builtin_amdgcn_s_wait_asynccnt(0);
#else
            const int nb = buf ^ 1;
            *(v8h*)&sA[nb][grow][gcol]     = a0;
            *(v8h*)&sA[nb][grow][gcol + 8] = a1;
            *(v8h*)&sW[nb][grow][gcol]     = w0;
            *(v8h*)&sW[nb][grow][gcol + 8] = w1;
#endif
            buf ^= 1;
        }
        __syncthreads();
    }

    // ---- epilogue: per frag, lane holds 8 vals; row = r + 8*(lane>>4)
    const int hi = lane >> 4;
    v8f accs[2][2] = {{c00, c01}, {c10, c11}};
#pragma unroll
    for (int ti = 0; ti < 2; ++ti) {
#pragma unroll
        for (int tj = 0; tj < 2; ++tj) {
            const v8f acc = accs[ti][tj];
            const int nc = n0 + wni + tj * 16 + (lane & 15);
#pragma unroll
            for (int r = 0; r < 8; ++r) {
                const int mr = m0 + wmi + ti * 16 + hi * 8 + r;
                float v = acc[r];
                if constexpr (EPI == EPI_F16) {
                    outH[(size_t)mr * ldo + nc] = (_Float16)v;
                } else if constexpr (EPI == EPI_XPROJ) {
                    // dbc split: [0,24) dt (padded to 32), [24,40) B, [40,56) C
                    if (nc < 32)
                        dtA[(size_t)mr * 32 + nc] =
                            (nc < DTRANK) ? (_Float16)v : (_Float16)0.f;
                    if (nc >= 24 && nc < 40)
                        BmO[(size_t)mr * DSTATE + (nc - 24)] = v;
                    else if (nc >= 40 && nc < 56)
                        CmO[(size_t)mr * DSTATE + (nc - 40)] = v;
                } else if constexpr (EPI == EPI_SOFTPLUS) {
                    v += bias[nc];
                    outF[(size_t)mr * ldo + nc] =
                        (v > 20.f) ? v : log1pf(expf(v));
                } else if constexpr (EPI == EPI_GELU_F16) {
                    v += bias[nc];
                    float g = 0.5f * v * (1.f + erff(v * 0.70710678118654752f));
                    outH[(size_t)mr * ldo + nc] = (_Float16)g;
                } else { // EPI_TRANS: (b,l,c) -> (b,c,h,w) f32 store
                    v += bias[nc];
                    const int b = mr >> 10, l = mr & 1023;
                    outF[((size_t)b * CCH + nc) * HW + l] = v;
                }
            }
        }
    }
}

// ---------------------------------------------------------------------------
// LayerNorm over C=384, reading NCHW x, writing f16 (B*L, C) row-major.
// ---------------------------------------------------------------------------
__global__ __launch_bounds__(256) void layernorm_f16(
    const float* __restrict__ x, const float* __restrict__ g,
    const float* __restrict__ bta, _Float16* __restrict__ out)
{
    const int row  = blockIdx.x * 8 + (threadIdx.x >> 5);
    const int lane = threadIdx.x & 31;
    const int b = row >> 10, l = row & 1023;
    const float* px = x + (size_t)b * CCH * HW + l;   // element c at px[c*HW]

    float s = 0.f, ss = 0.f;
    for (int c = lane; c < CCH; c += 32) {
        float v = px[(size_t)c * HW];
        s += v; ss += v * v;
    }
#pragma unroll
    for (int off = 16; off; off >>= 1) {
        s  += __shfl_xor(s,  off, 32);
        ss += __shfl_xor(ss, off, 32);
    }
    const float mu = s * (1.f / CCH);
    const float rs = rsqrtf(ss * (1.f / CCH) - mu * mu + 1e-5f);
    _Float16* po = out + (size_t)row * CCH;
    for (int c = lane; c < CCH; c += 32)
        po[c] = (_Float16)((px[(size_t)c * HW] - mu) * rs * g[c] + bta[c]);
}

// ---------------------------------------------------------------------------
// Causal depthwise conv (K=3) over L within each batch + SiLU.
// ---------------------------------------------------------------------------
__global__ __launch_bounds__(256) void conv_silu(
    const _Float16* __restrict__ xz, const float* __restrict__ cw,
    const float* __restrict__ cb, float* __restrict__ xcF,
    _Float16* __restrict__ xcH)
{
    const size_t i = (size_t)blockIdx.x * 256 + threadIdx.x;   // MROWS*DINNER
    const int ch = (int)(i % DINNER);
    const size_t row = i / DINNER;
    const int l = (int)(row & 1023);
    const _Float16* p = xz + row * (2 * DINNER) + ch;

    float v = (float)p[0] * cw[ch * 3 + 2];
    if (l >= 1) v += (float)p[-(2 * DINNER)] * cw[ch * 3 + 1];
    if (l >= 2) v += (float)p[-(4 * DINNER)] * cw[ch * 3 + 0];
    v += cb[ch];
    const float s = v / (1.f + __expf(-v));
    xcF[i] = s;
    xcH[i] = (_Float16)s;
}

// ---------------------------------------------------------------------------
// Selective scan: block = one batch, thread = one channel d (768 threads).
// ---------------------------------------------------------------------------
__global__ __launch_bounds__(768) void selective_scan(
    const float* __restrict__ dtf, const float* __restrict__ xcF,
    const _Float16* __restrict__ xz16, const float* __restrict__ Bm,
    const float* __restrict__ Cm, const float* __restrict__ A_log,
    const float* __restrict__ Dp, _Float16* __restrict__ ssm)
{
    const int b = blockIdx.x, d = threadIdx.x;
    __shared__ float sB[64][DSTATE];
    __shared__ float sC[64][DSTATE];

    float A[DSTATE], h[DSTATE];
#pragma unroll
    for (int n = 0; n < DSTATE; ++n) {
        A[n] = -__expf(A_log[(size_t)d * DSTATE + n]);
        h[n] = 0.f;
    }
    const float Dd = Dp[d];

    for (int l0 = 0; l0 < 1024; l0 += 64) {
        for (int i = threadIdx.x; i < 64 * DSTATE; i += 768) {
            const size_t g = ((size_t)(b * 1024 + l0) * DSTATE) + i;
            sB[i >> 4][i & 15] = Bm[g];
            sC[i >> 4][i & 15] = Cm[g];
        }
        __syncthreads();
        for (int l = 0; l < 64; ++l) {
            const size_t row = (size_t)b * 1024 + l0 + l;
            const float dt = dtf[row * DINNER + d];
            const float u  = xcF[row * DINNER + d];
            const float du = dt * u;
            float y = 0.f;
#pragma unroll
            for (int n = 0; n < DSTATE; ++n) {
                h[n] = __expf(dt * A[n]) * h[n] + du * sB[l][n];
                y += h[n] * sC[l][n];
            }
            y += u * Dd;
            const float z = (float)xz16[row * (2 * DINNER) + DINNER + d];
            y *= z / (1.f + __expf(-z));
            ssm[row * DINNER + d] = (_Float16)y;
        }
        __syncthreads();
    }
}

// ---------------------------------------------------------------------------
// f32 -> f16 weight conversion with zero padding to (drows, dcols).
// ---------------------------------------------------------------------------
__global__ __launch_bounds__(256) void cvt_pad(
    const float* __restrict__ src, _Float16* __restrict__ dst,
    int rows, int cols, int drows, int dcols)
{
    const int i = blockIdx.x * 256 + threadIdx.x;
    if (i >= drows * dcols) return;
    const int r = i / dcols, c = i % dcols;
    dst[i] = (r < rows && c < cols) ? (_Float16)src[(size_t)r * cols + c]
                                    : (_Float16)0.f;
}

// ---------------------------------------------------------------------------
extern "C" void kernel_launch(void* const* d_in, const int* in_sizes, int n_in,
                              void* d_out, int out_size, void* d_ws, size_t ws_size,
                              hipStream_t stream) {
    (void)in_sizes; (void)n_in; (void)out_size; (void)ws_size;
    const float* x      = (const float*)d_in[0];
    const float* ln_w   = (const float*)d_in[1];
    const float* ln_b   = (const float*)d_in[2];
    const float* in_w   = (const float*)d_in[3];   // (1536, 384)
    const float* conv_w = (const float*)d_in[4];   // (768, 3)
    const float* conv_b = (const float*)d_in[5];
    const float* xp_w   = (const float*)d_in[6];   // (56, 768)
    const float* dtp_w  = (const float*)d_in[7];   // (768, 24)
    const float* dtp_b  = (const float*)d_in[8];
    const float* A_log  = (const float*)d_in[9];   // (768, 16)
    const float* Dp     = (const float*)d_in[10];
    const float* out_w  = (const float*)d_in[11];  // (384, 768)
    const float* m1_w   = (const float*)d_in[12];  // (1536, 384)
    const float* m1_b   = (const float*)d_in[13];
    const float* m2_w   = (const float*)d_in[14];  // (384, 1536)
    const float* m2_b   = (const float*)d_in[15];
    float* out          = (float*)d_out;

    char* ws = (char*)d_ws;
    size_t off = 0;
    auto alloc = [&](size_t bytes) {
        size_t o = off; off += (bytes + 255) & ~(size_t)255; return o;
    };
    _Float16* w16_in  = (_Float16*)(ws + alloc((size_t)1536 * 384 * 2));
    _Float16* w16_xp  = (_Float16*)(ws + alloc((size_t)64   * 768 * 2));   // padded rows
    _Float16* w16_dt  = (_Float16*)(ws + alloc((size_t)768  * 32  * 2));   // padded cols
    _Float16* w16_out = (_Float16*)(ws + alloc((size_t)384  * 768 * 2));
    _Float16* w16_m1  = (_Float16*)(ws + alloc((size_t)1536 * 384 * 2));
    _Float16* w16_m2  = (_Float16*)(ws + alloc((size_t)384  * 1536 * 2));
    _Float16* t16     = (_Float16*)(ws + alloc((size_t)MROWS * CCH * 2));
    _Float16* xz16    = (_Float16*)(ws + alloc((size_t)MROWS * 2 * DINNER * 2));
    float*    xc32    = (float*)   (ws + alloc((size_t)MROWS * DINNER * 4));
    _Float16* xc16    = (_Float16*)(ws + alloc((size_t)MROWS * DINNER * 2));
    _Float16* dtA16   = (_Float16*)(ws + alloc((size_t)MROWS * 32 * 2));
    float*    Bm32    = (float*)   (ws + alloc((size_t)MROWS * DSTATE * 4));
    float*    Cm32    = (float*)   (ws + alloc((size_t)MROWS * DSTATE * 4));
    float*    dtfull  = (float*)   (ws + alloc((size_t)MROWS * DINNER * 4));
    _Float16* ssm16   = (_Float16*)(ws + alloc((size_t)MROWS * DINNER * 2));
    _Float16* t2_16   = (_Float16*)(ws + alloc((size_t)MROWS * CCH * 2));
    _Float16* h1_16   = xz16;  // alias: xz consumed before MLP hidden is written

    // --- weight conversion (f32 -> f16, zero-padded where needed) ---
    auto cvg = [](int n) { return dim3((n + 255) / 256); };
    cvt_pad<<<cvg(1536 * 384), 256, 0, stream>>>(in_w,  w16_in,  1536, 384, 1536, 384);
    cvt_pad<<<cvg(64 * 768),   256, 0, stream>>>(xp_w,  w16_xp,  56,   768, 64,   768);
    cvt_pad<<<cvg(768 * 32),   256, 0, stream>>>(dtp_w, w16_dt,  768,  24,  768,  32);
    cvt_pad<<<cvg(384 * 768),  256, 0, stream>>>(out_w, w16_out, 384,  768, 384,  768);
    cvt_pad<<<cvg(1536 * 384), 256, 0, stream>>>(m1_w,  w16_m1,  1536, 384, 1536, 384);
    cvt_pad<<<cvg(384 * 1536), 256, 0, stream>>>(m2_w,  w16_m2,  384,  1536, 384, 1536);

    // --- LayerNorm (NCHW -> (B*L, C) f16) ---
    layernorm_f16<<<dim3(MROWS / 8), 256, 0, stream>>>(x, ln_w, ln_b, t16);

    // --- in_proj: xz = t @ in_proj_w^T  (8192 x 384 x 1536), f16 out ---
    wmma_gemm<EPI_F16><<<dim3(1536 / 64, MROWS / 64), 128, 0, stream>>>(
        t16, CCH, w16_in, CCH, CCH,
        nullptr, xz16, 2 * DINNER, nullptr, nullptr, nullptr, nullptr);

    // --- causal depthwise conv + SiLU ---
    conv_silu<<<dim3((MROWS * DINNER) / 256), 256, 0, stream>>>(
        xz16, conv_w, conv_b, xc32, xc16);

    // --- x_proj: dbc = xc @ x_proj_w^T (8192 x 768 x 64pad), split epilogue ---
    wmma_gemm<EPI_XPROJ><<<dim3(1, MROWS / 64), 128, 0, stream>>>(
        xc16, DINNER, w16_xp, DINNER, DINNER,
        nullptr, nullptr, 0, nullptr, dtA16, Bm32, Cm32);

    // --- dt_proj + softplus: dt = softplus(dtA @ dt_proj_w^T + b) ---
    wmma_gemm<EPI_SOFTPLUS><<<dim3(DINNER / 64, MROWS / 64), 128, 0, stream>>>(
        dtA16, 32, w16_dt, 32, 32,
        dtfull, nullptr, DINNER, dtp_b, nullptr, nullptr, nullptr);

    // --- selective scan + D skip + SiLU(z) gate ---
    selective_scan<<<dim3(BATCH), 768, 0, stream>>>(
        dtfull, xc32, xz16, Bm32, Cm32, A_log, Dp, ssm16);

    // --- out_proj: t2 = y @ out_proj_w^T (8192 x 768 x 384), f16 out ---
    wmma_gemm<EPI_F16><<<dim3(CCH / 64, MROWS / 64), 128, 0, stream>>>(
        ssm16, DINNER, w16_out, DINNER, DINNER,
        nullptr, t2_16, CCH, nullptr, nullptr, nullptr, nullptr);

    // --- MLP fc1 + exact GELU (8192 x 384 x 1536), f16 out ---
    wmma_gemm<EPI_GELU_F16><<<dim3(1536 / 64, MROWS / 64), 128, 0, stream>>>(
        t2_16, CCH, w16_m1, CCH, CCH,
        nullptr, h1_16, 1536, m1_b, nullptr, nullptr, nullptr);

    // --- MLP fc2 + bias + NHWC->NCHW transpose into d_out (f32) ---
    wmma_gemm<EPI_TRANS><<<dim3(CCH / 64, MROWS / 64), 128, 0, stream>>>(
        h1_16, 1536, w16_m2, 1536, 1536,
        out, nullptr, CCH, m2_b, nullptr, nullptr, nullptr);
}
